// GNNFewShotClassifier_71236327571546
// MI455X (gfx1250) — matrix operand order repaired and checked by reading, not compiled, not run on hardware
//
#include <hip/hip_runtime.h>
#include <hip/hip_bf16.h>
#include <math.h>

typedef __attribute__((ext_vector_type(16))) _Float16 v16h;
typedef __attribute__((ext_vector_type(8)))  _Float16 v8h;
typedef __attribute__((ext_vector_type(8)))  float    v8f;

union V16U { v16h v; v8h h[2]; };

#define S_SUP 100
#define DIM   512
#define H1    256
#define H2    128
#define NQ    2048
#define NC    20

// workspace byte offsets
#define WS_T    0                         // 20*256 f32 (layer1 class constants, pre-relu)
#define WS_GW2  (WS_T + NC*H1*4)          // 128 f32 (g@W2 + b2)
#define WS_W1HT (WS_GW2 + H2*4)           // 256*512 f16, W1hT[n][k] = W1[k][n]
#define WS_W2HT (WS_W1HT + H1*DIM*2)      // 128*256 f16, W2hT[n][k] = W2[k][n]
#define WS_P    (WS_W2HT + H2*H1*2)       // 2048*256 f32  (X @ W1, unscaled)

__device__ __forceinline__ float k_alpha() { return rsqrtf(100.0f + 1e-6f) * rsqrtf(6.0f + 1e-6f); }
__device__ __forceinline__ float k_dS2()   { return 1.0f / (6.0f + 1e-6f); }

__device__ __forceinline__ v8f wmma16(v16h a, v16h b, v8f c) {
  return __builtin_amdgcn_wmma_f32_16x16x32_f16(false, a, false, b, (short)0, c, false, false);
}

// ------------------------------------------- converts (both transposed)
__global__ __launch_bounds__(256) void k_convert(const float* __restrict__ W1,
                                                 const float* __restrict__ W2,
                                                 _Float16* __restrict__ W1hT,
                                                 _Float16* __restrict__ W2hT) {
  int idx = blockIdx.x * 256 + threadIdx.x;
  if (idx < H1 * DIM) {
    int n = idx >> 9;            // DIM == 512
    int k = idx & (DIM - 1);
    W1hT[idx] = (_Float16)W1[(size_t)k * H1 + n];
  }
  int idx2 = idx - H1 * DIM;
  if (idx2 >= 0 && idx2 < H2 * H1) {
    int n = idx2 >> 8;           // H1 == 256
    int k = idx2 & (H1 - 1);
    W2hT[idx2] = (_Float16)W2[(size_t)k * H2 + n];
  }
}

// ------------------------------------------------------- graph constants
__global__ __launch_bounds__(256) void k_consts(const float* __restrict__ sf,
                                                const int* __restrict__ lab,
                                                const float* __restrict__ W1,
                                                const float* __restrict__ b1,
                                                const float* __restrict__ W2,
                                                const float* __restrict__ b2,
                                                float* __restrict__ ws_t,
                                                float* __restrict__ ws_gw2) {
  __shared__ float csum[NC * DIM];   // 40 KB
  __shared__ float sall[DIM];
  __shared__ float h10[H1];
  const int tid = threadIdx.x;
  const float dS2 = k_dS2(), al = k_alpha();

  for (int d = tid; d < DIM; d += 256) {
    float tot = 0.f;
    for (int s = 0; s < S_SUP; ++s) tot += sf[(size_t)s * DIM + d];
    sall[d] = tot;
  }
  for (int o = tid; o < NC * DIM; o += 256) {
    int c = o / DIM, d = o % DIM;
    float a = 0.f;
    for (int s = 0; s < S_SUP; ++s) a += (lab[s] == c) ? sf[(size_t)s * DIM + d] : 0.f;
    csum[o] = a;
  }
  __syncthreads();

  // t_c = dS2 * classSum_c @ W1 + b1   (pre-activation constant per class)
  for (int o = tid; o < NC * H1; o += 256) {
    int c = o >> 8, j = o & (H1 - 1);
    float z = b1[j];
    for (int d = 0; d < DIM; ++d) z += dS2 * csum[c * DIM + d] * W1[d * H1 + j];
    ws_t[o] = z;
  }
  // h1_0 = relu(alpha * sumAll @ W1 + b1)   (query-node layer-1, constant)
  if (tid < H1) {
    float z = b1[tid];
    for (int d = 0; d < DIM; ++d) z += al * sall[d] * W1[d * H1 + tid];
    h10[tid] = fmaxf(z, 0.f);
  }
  __syncthreads();
  // gw2 = (alpha*h1_0) @ W2 + b2
  if (tid < H2) {
    float z = b2[tid];
    for (int j = 0; j < H1; ++j) z += al * h10[j] * W2[j * H2 + tid];
    ws_gw2[tid] = z;
  }
}

// ------------------------------------------- GEMM1: P = X @ W1 (f16 WMMA)
__global__ __launch_bounds__(256) void k_gemm1(const float* __restrict__ X,
                                               const _Float16* __restrict__ W1hT,
                                               float* __restrict__ P) {
  __shared__ _Float16 As[64 * 32];    // row-major M x K chunk (f32->f16 staged)
  const int tid = threadIdx.x;
  const int wave = tid >> 5, lane = tid & 31;
  const int mtile = wave >> 1;          // 0..3
  const int nbase = (wave & 1) * 2;     // tiles nbase, nbase+1
  const int half = lane >> 4;           // 0/1
  const int mn = lane & 15;
  const int mblk = blockIdx.x * 64;
  const int nblk = blockIdx.y * 64;
  // B fragment base pointers (contiguous 16B chunks from pre-transposed W1hT)
  const _Float16* gB0 = W1hT + (size_t)(nblk + (nbase + 0) * 16 + mn) * DIM + half * 16;
  const _Float16* gB1 = W1hT + (size_t)(nblk + (nbase + 1) * 16 + mn) * DIM + half * 16;
  v8f acc0 = {}; v8f acc1 = {};

  for (int k0 = 0; k0 < DIM; k0 += 32) {
    __syncthreads();
    { // stage A with f32->f16 conversion (64x32)
      int row = tid >> 2, seg = (tid & 3) * 8;
      const float* src = X + (size_t)(mblk + row) * DIM + k0 + seg;
      v8h tmp;
      #pragma unroll
      for (int i = 0; i < 8; ++i) tmp[i] = (_Float16)src[i];
      *(v8h*)&As[row * 32 + seg] = tmp;
    }
    __syncthreads();
    V16U a;
    a.h[0] = *(const v8h*)&As[(mtile * 16 + mn) * 32 + half * 16];
    a.h[1] = *(const v8h*)&As[(mtile * 16 + mn) * 32 + half * 16 + 8];
    V16U fb0, fb1;
    fb0.h[0] = *(const v8h*)(gB0 + k0);
    fb0.h[1] = *(const v8h*)(gB0 + k0 + 8);
    fb1.h[0] = *(const v8h*)(gB1 + k0);
    fb1.h[1] = *(const v8h*)(gB1 + k0 + 8);
    acc0 = wmma16(a.v, fb0.v, acc0);
    acc1 = wmma16(a.v, fb1.v, acc1);
  }
  const int m0 = mblk + mtile * 16 + half * 8;
  const int n0a = nblk + (nbase + 0) * 16 + mn;
  const int n0b = nblk + (nbase + 1) * 16 + mn;
  #pragma unroll
  for (int r = 0; r < 8; ++r) {
    P[(size_t)(m0 + r) * H1 + n0a] = acc0[r];
    P[(size_t)(m0 + r) * H1 + n0b] = acc1[r];
  }
}

// ------------- fused per-query: U build, V = U@W2 (WMMA), cosine epilogue
__global__ __launch_bounds__(256) void k_classify(const float* __restrict__ P,
                                                  const float* __restrict__ ws_t,
                                                  const float* __restrict__ ws_gw2,
                                                  const _Float16* __restrict__ W2hT,
                                                  const float* __restrict__ b2,
                                                  float* __restrict__ out) {
  __shared__ _Float16 Uh[32 * H1];   // 16 KB, classes padded to 32 rows
  __shared__ float Vs[NC * H2];      // 10 KB
  __shared__ float QE[H2];
  const int tid = threadIdx.x;
  const int wave = tid >> 5, lane = tid & 31;
  const int mtile = wave >> 2;           // 0..1
  const int nbase = (wave & 3) * 2;      // 0,2,4,6
  const int half = lane >> 4;
  const int mn = lane & 15;
  const float al = k_alpha();
  const float s56 = 5.0f * k_dS2();
  const float s5a = 5.0f * al;
  const _Float16* gB0 = W2hT + (size_t)((nbase + 0) * 16 + mn) * H1 + half * 16;
  const _Float16* gB1 = W2hT + (size_t)((nbase + 1) * 16 + mn) * H1 + half * 16;

  for (int qi = 0; qi < 4; ++qi) {
    const int q = blockIdx.x * 4 + qi;
    // U rows: u_c = relu(alpha*P[q] + t_c), rows 20..31 zero-padded
    for (int idx = tid; idx < 32 * H1; idx += 256) {
      int c = idx >> 8, j = idx & (H1 - 1);
      float v = 0.f;
      if (c < NC) v = fmaxf(al * P[(size_t)q * H1 + j] + ws_t[c * H1 + j], 0.f);
      Uh[idx] = (_Float16)v;
    }
    __syncthreads();

    v8f acc0 = {}; v8f acc1 = {};
    for (int k0 = 0; k0 < H1; k0 += 32) {
      V16U a;
      a.h[0] = *(const v8h*)&Uh[(mtile * 16 + mn) * H1 + k0 + half * 16];
      a.h[1] = *(const v8h*)&Uh[(mtile * 16 + mn) * H1 + k0 + half * 16 + 8];
      V16U fb0, fb1;
      fb0.h[0] = *(const v8h*)(gB0 + k0);
      fb0.h[1] = *(const v8h*)(gB0 + k0 + 8);
      fb1.h[0] = *(const v8h*)(gB1 + k0);
      fb1.h[1] = *(const v8h*)(gB1 + k0 + 8);
      acc0 = wmma16(a.v, fb0.v, acc0);
      acc1 = wmma16(a.v, fb1.v, acc1);
    }
    const int m0 = mtile * 16 + half * 8;
    #pragma unroll
    for (int r = 0; r < 8; ++r) {
      int row = m0 + r;
      if (row < NC) {
        Vs[row * H2 + (nbase + 0) * 16 + mn] = acc0[r];
        Vs[row * H2 + (nbase + 1) * 16 + mn] = acc1[r];
      }
    }
    __syncthreads();
    // q_emb = relu(5*alpha * sum_c V_c + b2)
    if (tid < H2) {
      float s = 0.f;
      #pragma unroll
      for (int c = 0; c < NC; ++c) s += Vs[c * H2 + tid];
      QE[tid] = fmaxf(s5a * s + b2[tid], 0.f);
    }
    __syncthreads();
    // cosine(q_emb, proto_c), proto_c = relu(gw2 + (5/(6+eps)) V_c)
    for (int c = wave; c < NC; c += 8) {
      float dot = 0.f, pn = 0.f, qn = 0.f;
      for (int n = lane; n < H2; n += 32) {
        float qe = QE[n];
        float pc = fmaxf(ws_gw2[n] + s56 * Vs[c * H2 + n], 0.f);
        dot += qe * pc; pn += pc * pc; qn += qe * qe;
      }
      #pragma unroll
      for (int off = 16; off > 0; off >>= 1) {
        dot += __shfl_xor(dot, off, 32);
        pn  += __shfl_xor(pn,  off, 32);
        qn  += __shfl_xor(qn,  off, 32);
      }
      if (lane == 0)
        out[(size_t)q * NC + c] = dot / fmaxf(sqrtf(qn) * sqrtf(pn), 1e-8f);
    }
    __syncthreads();
  }
}

extern "C" void kernel_launch(void* const* d_in, const int* in_sizes, int n_in,
                              void* d_out, int out_size, void* d_ws, size_t ws_size,
                              hipStream_t stream) {
  (void)in_sizes; (void)n_in; (void)out_size; (void)ws_size;
  const float* sf  = (const float*)d_in[0];
  const int*   lab = (const int*)d_in[1];
  const float* X   = (const float*)d_in[2];
  const float* W1  = (const float*)d_in[3];
  const float* b1  = (const float*)d_in[4];
  const float* W2  = (const float*)d_in[5];
  const float* b2  = (const float*)d_in[6];
  float* out = (float*)d_out;

  char* ws = (char*)d_ws;
  float*    ws_t   = (float*)(ws + WS_T);
  float*    ws_gw2 = (float*)(ws + WS_GW2);
  _Float16* W1hT   = (_Float16*)(ws + WS_W1HT);
  _Float16* W2hT   = (_Float16*)(ws + WS_W2HT);
  float*    Pm     = (float*)(ws + WS_P);

  k_convert<<<(H1 * DIM + H2 * H1 + 255) / 256, 256, 0, stream>>>(W1, W2, W1hT, W2hT);
  k_consts<<<1, 256, 0, stream>>>(sf, lab, W1, b1, W2, b2, ws_t, ws_gw2);
  k_gemm1<<<dim3(NQ / 64, H1 / 64), 256, 0, stream>>>(X, W1hT, Pm);
  k_classify<<<NQ / 4, 256, 0, stream>>>(Pm, ws_t, ws_gw2, W2hT, b2, out);
}
